// DeltaNet_31877247271507
// MI455X (gfx1250) — compile-verified, matrix operand order checked by reading, not run on hardware
//
#include <hip/hip_runtime.h>
#include <math.h>

typedef __attribute__((ext_vector_type(16))) _Float16 v16h;
typedef __attribute__((ext_vector_type(8)))  _Float16 v8h;
typedef __attribute__((ext_vector_type(8)))  float    v8f;

#define WMMA(a,b,c) __builtin_amdgcn_wmma_f32_16x16x32_f16(false,(a),false,(b),(short)0,(c),false,false)

static constexpr int B   = 2;
static constexpr int L   = 2048;
static constexpr int D   = 1024;
static constexpr int H   = 4;
static constexpr int DH  = 256;    // head dim (Dk == Dv)
static constexpr int CH  = 32;     // delta chunk
static constexpr int M   = B * L;  // 4096 token rows
static constexpr int HID = 2048;
static constexpr int GIN = 1056;

static __device__ __forceinline__ v8f vzero() {
  v8f z = {0.f,0.f,0.f,0.f,0.f,0.f,0.f,0.f};
  return z;
}

// A/B fragment from a row-major row pointer: elements 0..7 = k[kpos..kpos+7],
// elements 8..15 = k[kpos+16..kpos+23]  (caller passes kpos = k0 + kofs)
static __device__ __forceinline__ v16h frag_row(const _Float16* p0, int kpos) {
  const _Float16* p = p0 + kpos;
  v8h lo = *(const v8h*)p;
  v8h hi = *(const v8h*)(p + 16);
  v16h r;
  #pragma unroll
  for (int e = 0; e < 8; ++e) { r[e] = lo[e]; r[e + 8] = hi[e]; }
  return r;
}

// Fragment gathering a *column* of a row-major matrix (strided): used for
// B-operands whose k-dim runs down storage rows, and for k^T A-operands.
static __device__ __forceinline__ v16h frag_col(const _Float16* base, int ld, int col, int kofs) {
  v16h r;
  #pragma unroll
  for (int e = 0; e < 16; ++e) {
    int k = kofs + ((e < 8) ? e : (e + 8));
    r[e] = base[(size_t)k * ld + col];
  }
  return r;
}

// Lane-local retype of two stacked f32 C-layout accumulator tiles (rows 0..15
// and 16..31 of one 16-col slice) into a f16 B-operand fragment (k=32).
static __device__ __forceinline__ v16h accfragB(const v8f& t0, const v8f& t1) {
  v16h r;
  #pragma unroll
  for (int e = 0; e < 8; ++e) { r[e] = (_Float16)t0[e]; r[e + 8] = (_Float16)t1[e]; }
  return r;
}

// CDNA5 async memory->LDS copy (ASYNCcnt-tracked), GV addressing mode.
static __device__ __forceinline__ void async_ld_lds_b128(unsigned lds_byte_addr,
                                                         const void* gaddr) {
  asm volatile("global_load_async_to_lds_b128 %0, %1, off"
               :: "v"(lds_byte_addr), "v"(gaddr) : "memory");
}
static __device__ __forceinline__ void wait_asynccnt0() {
  asm volatile("s_wait_asynccnt 0x0" ::: "memory");
}

// ---------------------------------------------------------------- f32 -> f16
__global__ void k_cvt_f16(const float* __restrict__ s, _Float16* __restrict__ d, long long n) {
  long long i = (long long)blockIdx.x * blockDim.x + threadIdx.x;
  long long st = (long long)gridDim.x * blockDim.x;
  for (; i < n; i += st) d[i] = (_Float16)s[i];
}

// --------------- Y(MxN) = X(MxK) @ W(NxK)^T, 64x64 tile per wave (16 WMMAs/k)
__global__ __launch_bounds__(32) void k_gemm64(const _Float16* __restrict__ X,
                                               const _Float16* __restrict__ W,
                                               float* __restrict__ Y,
                                               int Mn, int Nn, int Kn) {
  const int lane = threadIdx.x;
  const int frow = lane & 15;
  const int kofs = (lane < 16) ? 0 : 8;
  const int n0 = blockIdx.x * 64, m0 = blockIdx.y * 64;
  const _Float16* xr[4];
  const _Float16* wr[4];
  #pragma unroll
  for (int i = 0; i < 4; ++i) {
    xr[i] = X + (size_t)(m0 + i * 16 + frow) * Kn;
    wr[i] = W + (size_t)(n0 + i * 16 + frow) * Kn;
  }
  v8f acc[16];
  #pragma unroll
  for (int i = 0; i < 16; ++i) acc[i] = vzero();

  for (int k0 = 0; k0 < Kn; k0 += 32) {
    __builtin_prefetch(xr[0] + k0 + 512, 0, 0);
    __builtin_prefetch(wr[0] + k0 + 512, 0, 0);
    v16h a[4], b[4];
    #pragma unroll
    for (int i = 0; i < 4; ++i) {
      a[i] = frag_row(xr[i], k0 + kofs);
      b[i] = frag_row(wr[i], k0 + kofs);
    }
    #pragma unroll
    for (int i = 0; i < 4; ++i)
      #pragma unroll
      for (int j = 0; j < 4; ++j)
        acc[i * 4 + j] = WMMA(a[i], b[j], acc[i * 4 + j]);
  }
  const int col = lane & 15, rb = (lane >> 4) * 8;
  #pragma unroll
  for (int i = 0; i < 4; ++i)
    #pragma unroll
    for (int j = 0; j < 4; ++j)
      #pragma unroll
      for (int r = 0; r < 8; ++r)
        Y[(size_t)(m0 + i * 16 + rb + r) * Nn + n0 + j * 16 + col] = acc[i * 4 + j][r];
}

// ------------------------- 16-wide variant (for the N=16 logits projection)
__global__ __launch_bounds__(32) void k_gemm16(const _Float16* __restrict__ X,
                                               const _Float16* __restrict__ W,
                                               float* __restrict__ Y,
                                               int Mn, int Nn, int Kn) {
  const int lane = threadIdx.x;
  const int frow = lane & 15;
  const int kofs = (lane < 16) ? 0 : 8;
  const int n0 = blockIdx.x * 16, m0 = blockIdx.y * 16;
  const _Float16* xr = X + (size_t)(m0 + frow) * Kn;
  const _Float16* wr = W + (size_t)(n0 + frow) * Kn;
  v8f acc = vzero();
  for (int k0 = 0; k0 < Kn; k0 += 32) {
    __builtin_prefetch(xr + k0 + 512, 0, 0);
    v16h a = frag_row(xr, k0 + kofs);
    v16h b = frag_row(wr, k0 + kofs);
    acc = WMMA(a, b, acc);
  }
  const int col = lane & 15, rb = (lane >> 4) * 8;
  #pragma unroll
  for (int r = 0; r < 8; ++r)
    Y[(size_t)(m0 + rb + r) * Nn + n0 + col] = acc[r];
}

// --------------------------------------- depthwise causal conv K=4 then SiLU
__global__ void k_conv4_silu(const float* __restrict__ x, const float* __restrict__ w,
                             float* __restrict__ y) {
  long long i = (long long)blockIdx.x * blockDim.x + threadIdx.x;
  long long st = (long long)gridDim.x * blockDim.x;
  for (; i < (long long)M * D; i += st) {
    int d = (int)(i % D);
    long long bl = i / D;
    int l = (int)(bl % L);
    long long b = bl / L;
    float s = 0.f;
    #pragma unroll
    for (int t = 0; t < 4; ++t) {
      int ls = l + t - 3;
      if (ls >= 0) s += x[((size_t)b * L + ls) * D + d] * w[d * 4 + t];
    }
    y[i] = s / (1.f + expf(-s));   // silu
  }
}

// -------------------------------------------- per-(head,chan) causal FIR (K)
__global__ void k_fir(const float* __restrict__ x, const float* __restrict__ w,
                      float* __restrict__ y, int K) {
  long long i = (long long)blockIdx.x * blockDim.x + threadIdx.x;
  long long st = (long long)gridDim.x * blockDim.x;
  for (; i < (long long)M * D; i += st) {
    int d = (int)(i % D);
    long long bl = i / D;
    int l = (int)(bl % L);
    long long b = bl / L;
    float s = 0.f;
    for (int t = 0; t < K; ++t) {
      int ls = l + t - (K - 1);
      if (ls >= 0) s += x[((size_t)b * L + ls) * D + d] * w[(size_t)d * K + t];
    }
    y[i] = s;
  }
}

// ------------------------------------------------- beta = sigmoid(hs @ Wb^T)
__global__ __launch_bounds__(256) void k_beta(const float* __restrict__ hs,
                                              const float* __restrict__ Wb,
                                              float* __restrict__ beta) {
  int gw = blockIdx.x * 8 + (threadIdx.x >> 5);  // over M*H
  int lane = threadIdx.x & 31;
  int h = gw % H;
  int bl = gw / H;
  const float* xr = hs + (size_t)bl * D;
  const float* wr = Wb + (size_t)h * D;
  float s = 0.f;
  for (int d = lane; d < D; d += 32) s += xr[d] * wr[d];
  #pragma unroll
  for (int o = 16; o; o >>= 1) s += __shfl_xor(s, o);
  if (lane == 0) beta[gw] = 1.f / (1.f + expf(-s));
}

// ---- l2norm(q), l2norm(k), k*beta, v*beta  -> f16, layout [b*H+h][l][dh]
__global__ __launch_bounds__(256) void k_prep(const float* __restrict__ q,
                                              const float* __restrict__ k,
                                              const float* __restrict__ v,
                                              const float* __restrict__ beta,
                                              _Float16* __restrict__ qn,
                                              _Float16* __restrict__ kn,
                                              _Float16* __restrict__ kb,
                                              _Float16* __restrict__ vb) {
  int gw = blockIdx.x * 8 + (threadIdx.x >> 5);  // over M*H
  int lane = threadIdx.x & 31;
  int h = gw % H;
  int bl = gw / H;
  int b = bl / L, l = bl % L;
  size_t src = (size_t)bl * D + h * DH;
  size_t dst = ((size_t)(b * H + h) * L + l) * DH;
  float bet = beta[gw];
  float qv[8], kv[8];
  float sq = 0.f, sk = 0.f;
  #pragma unroll
  for (int j = 0; j < 8; ++j) {
    int d = lane * 8 + j;
    qv[j] = q[src + d]; kv[j] = k[src + d];
    sq += qv[j] * qv[j]; sk += kv[j] * kv[j];
  }
  #pragma unroll
  for (int o = 16; o; o >>= 1) { sq += __shfl_xor(sq, o); sk += __shfl_xor(sk, o); }
  float rq = rsqrtf(sq + 1e-6f), rk = rsqrtf(sk + 1e-6f);
  #pragma unroll
  for (int j = 0; j < 8; ++j) {
    int d = lane * 8 + j;
    qn[dst + d] = (_Float16)(qv[j] * rq);
    float kkn = kv[j] * rk;
    kn[dst + d] = (_Float16)kkn;
    kb[dst + d] = (_Float16)(kkn * bet);
    vb[dst + d] = (_Float16)(v[src + d] * bet);
  }
}

// ------------------------------------------------------------- delta rule
// grid (B*H, DH/64), block 128 (4 waves). Each wave owns a 256x16 slice of S
// kept in 16 WMMA f32 accumulators; Dk-side chunk work shared through LDS.
// Chunk tiles are staged memory->LDS on the CDNA5 async engine (ASYNCcnt).
__global__ __launch_bounds__(128) void k_delta(const _Float16* __restrict__ qn,
                                               const _Float16* __restrict__ kn,
                                               const _Float16* __restrict__ kbq,
                                               const _Float16* __restrict__ vbq,
                                               float* __restrict__ out) {
  __shared__ __align__(16) _Float16 sK [CH * DH];   // 16 KB
  __shared__ __align__(16) _Float16 sKB[CH * DH];   // 16 KB
  __shared__ __align__(16) _Float16 sW [CH * DH];   // 16 KB
  __shared__ __align__(16) float    sA [CH * CH];   //  4 KB
  __shared__ __align__(16) _Float16 sT [CH * CH];   //  2 KB
  __shared__ __align__(16) _Float16 sAt[CH * CH];   //  2 KB

  const int tid  = threadIdx.x;
  const int lane = tid & 31;
  const int wave = tid >> 5;
  const int bh   = blockIdx.x;
  const int b    = bh / H, h = bh % H;
  const int dv0  = blockIdx.y * 64 + wave * 16;
  const size_t gbase = (size_t)bh * L * DH;
  const _Float16* gq  = qn  + gbase;
  const _Float16* gk  = kn  + gbase;
  const _Float16* gkb = kbq + gbase;
  const _Float16* gv  = vbq + gbase;

  const int frow = lane & 15;
  const int kofs = (lane < 16) ? 0 : 8;
  const int crow = lane & 15;
  const int crb  = (lane >> 4) * 8;

  const unsigned ldsK  = (unsigned)(size_t)(void*)sK;
  const unsigned ldsKB = (unsigned)(size_t)(void*)sKB;

  v8f S[16];
  #pragma unroll
  for (int i = 0; i < 16; ++i) S[i] = vzero();

  for (int c = 0; c < L / CH; ++c) {
    const size_t coff = (size_t)c * CH * DH;

    { // stage k, kb chunk tiles to LDS via async engine (128B per thread each)
      const char* srcK  = (const char*)(gk  + coff);
      const char* srcKB = (const char*)(gkb + coff);
      #pragma unroll
      for (int j = 0; j < 8; ++j) {
        const int byt = (tid + 128 * j) * 16;
        async_ld_lds_b128(ldsK  + byt, srcK  + byt);
        async_ld_lds_b128(ldsKB + byt, srcKB + byt);
      }
      wait_asynccnt0();
    }

    // A = strict_lower(-kb @ k^T), attn = lower_incl(q @ k^T) — 4 tiles/4 waves
    {
      const int tm = wave >> 1, tn = wave & 1;
      v8f accA = vzero(), accQ = vzero();
      #pragma unroll
      for (int kk = 0; kk < DH; kk += 32) {
        v16h bfr = frag_row(gk  + coff + (size_t)(tn * 16 + frow) * DH, kk + kofs);
        v16h aKB = frag_row(gkb + coff + (size_t)(tm * 16 + frow) * DH, kk + kofs);
        v16h aQ  = frag_row(gq  + coff + (size_t)(tm * 16 + frow) * DH, kk + kofs);
        accA = WMMA(aKB, bfr, accA);
        accQ = WMMA(aQ,  bfr, accQ);
      }
      #pragma unroll
      for (int r = 0; r < 8; ++r) {
        int gm = tm * 16 + crb + r, gn = tn * 16 + crow;
        sA [gm * CH + gn] = (gn <  gm) ? -accA[r] : 0.f;
        sAt[gm * CH + gn] = (gn <= gm) ? (_Float16)accQ[r] : (_Float16)0.f;
      }
    }
    __syncthreads();

    // serial triangular inversion (I + A)^-1 style row recurrence, wave 0
    if (wave == 0) {
      for (int i = 1; i < CH; ++i) {
        float rowv = sA[i * CH + lane];
        float add = 0.f;
        for (int j = 1; j < i; ++j)
          add += __shfl(rowv, j) * sA[j * CH + lane];  // zeros kill invalid terms
        sA[i * CH + lane] = rowv + add;
      }
    }
    __syncthreads();

    for (int p = tid; p < CH * CH; p += 128) {
      int r = p >> 5, cc = p & 31;
      sT[p] = (_Float16)(sA[p] + ((r == cc) ? 1.f : 0.f));
    }
    __syncthreads();

    // w = T @ kb  (32x256): 32 tiles over 4 waves
    #pragma unroll
    for (int s = 0; s < 8; ++s) {
      int t  = s * 4 + wave;
      int tm = t & 1, tn = t >> 1;
      v16h a  = frag_row(sT + (tm * 16 + frow) * CH, kofs);
      v16h bf = frag_col(sKB, DH, tn * 16 + frow, kofs);
      v8f acc = vzero();
      acc = WMMA(a, bf, acc);
      #pragma unroll
      for (int r = 0; r < 8; ++r)
        sW[(tm * 16 + crb + r) * DH + tn * 16 + crow] = (_Float16)acc[r];
    }

    // u = T @ v  (this wave's 16 cols of v)
    v8f u0, u1;
    {
      v16h bf = frag_col(gv + coff, DH, dv0 + frow, kofs);
      v16h a0 = frag_row(sT + (frow) * CH, kofs);
      v16h a1 = frag_row(sT + (16 + frow) * CH, kofs);
      v8f z0 = vzero(), z1 = vzero();
      u0 = WMMA(a0, bf, z0);
      u1 = WMMA(a1, bf, z1);
    }
    __syncthreads();   // sW visible to all waves

    // u_i = u - w @ S   (S consumed as B-operand by lane-local retype)
    v8f ws0 = vzero(), ws1 = vzero();
    #pragma unroll
    for (int kb2 = 0; kb2 < 8; ++kb2) {
      v16h bS = accfragB(S[2 * kb2], S[2 * kb2 + 1]);
      v16h a0 = frag_row(sW + (frow) * DH, kb2 * 32 + kofs);
      v16h a1 = frag_row(sW + (16 + frow) * DH, kb2 * 32 + kofs);
      ws0 = WMMA(a0, bS, ws0);
      ws1 = WMMA(a1, bS, ws1);
    }
    v8f ui0, ui1;
    #pragma unroll
    for (int e = 0; e < 8; ++e) { ui0[e] = u0[e] - ws0[e]; ui1[e] = u1[e] - ws1[e]; }

    // o = q @ S + attn @ u_i
    v8f o0 = vzero(), o1 = vzero();
    #pragma unroll
    for (int kb2 = 0; kb2 < 8; ++kb2) {
      v16h bS = accfragB(S[2 * kb2], S[2 * kb2 + 1]);
      v16h a0 = frag_row(gq + coff + (size_t)(frow) * DH, kb2 * 32 + kofs);
      v16h a1 = frag_row(gq + coff + (size_t)(16 + frow) * DH, kb2 * 32 + kofs);
      o0 = WMMA(a0, bS, o0);
      o1 = WMMA(a1, bS, o1);
    }
    {
      v16h bU = accfragB(ui0, ui1);
      v16h a0 = frag_row(sAt + (frow) * CH, kofs);
      v16h a1 = frag_row(sAt + (16 + frow) * CH, kofs);
      o0 = WMMA(a0, bU, o0);
      o1 = WMMA(a1, bU, o1);
      // S += k^T @ u_i  (k^T A-fragments gathered from LDS)
      #pragma unroll
      for (int i = 0; i < 16; ++i) {
        v16h a = frag_col(sK, DH, i * 16 + frow, kofs);
        S[i] = WMMA(a, bU, S[i]);
      }
    }

    // write o to (B,L,H,DH)
    #pragma unroll
    for (int r = 0; r < 8; ++r) {
      int l0 = c * CH + crb + r;
      int l1 = l0 + 16;
      out[((size_t)(b * L + l0) * H + h) * DH + dv0 + crow] = o0[r];
      out[((size_t)(b * L + l1) * H + h) * DH + dv0 + crow] = o1[r];
    }
    __syncthreads();   // protect LDS tiles before next chunk
  }
}

// --------------------------------- hidden -> gate_in[:,0:1024] (f32 + f16)
__global__ void k_copyhid(const float* __restrict__ hs, float* __restrict__ gin,
                          _Float16* __restrict__ gin16) {
  long long i = (long long)blockIdx.x * blockDim.x + threadIdx.x;
  long long st = (long long)gridDim.x * blockDim.x;
  for (; i < (long long)M * D; i += st) {
    int d = (int)(i % D);
    long long bl = i / D;
    float v = hs[i];
    gin[(size_t)bl * GIN + d] = v;
    gin16[(size_t)bl * GIN + d] = (_Float16)v;
  }
}

// -------------------- per-(token,path,head) mean/var -> gate_in[:,1024:1056]
__global__ __launch_bounds__(256) void k_stats(const float* __restrict__ p0,
                                               const float* __restrict__ p1,
                                               const float* __restrict__ p2,
                                               const float* __restrict__ p3,
                                               float* __restrict__ gin,
                                               _Float16* __restrict__ gin16) {
  int gw = blockIdx.x * 8 + (threadIdx.x >> 5);  // over M*4*H
  int lane = threadIdx.x & 31;
  int h = gw % H;
  int p = (gw / H) & 3;
  int bl = gw / (4 * H);
  const float* P = (p == 0) ? p0 : (p == 1) ? p1 : (p == 2) ? p2 : p3;
  size_t src = (size_t)bl * D + h * DH;
  float s = 0.f, s2 = 0.f;
  for (int d = lane; d < DH; d += 32) { float x = P[src + d]; s += x; s2 += x * x; }
  #pragma unroll
  for (int o = 16; o; o >>= 1) { s += __shfl_xor(s, o); s2 += __shfl_xor(s2, o); }
  if (lane == 0) {
    float m = s / (float)DH;
    float va = s2 / (float)DH - m * m;
    size_t o = (size_t)bl * GIN + D + p * 8 + h * 2;
    gin[o] = m; gin[o + 1] = va;
    gin16[o] = (_Float16)m; gin16[o + 1] = (_Float16)va;
  }
}

// ------------------------------------------------ bias + exact GeLU -> f16
__global__ void k_gelu(const float* __restrict__ x, const float* __restrict__ b1,
                       _Float16* __restrict__ y) {
  long long i = (long long)blockIdx.x * blockDim.x + threadIdx.x;
  long long st = (long long)gridDim.x * blockDim.x;
  for (; i < (long long)M * HID; i += st) {
    float v = x[i] + b1[i % HID];
    float g = 0.5f * v * (1.f + erff(v * 0.70710678118f));
    y[i] = (_Float16)g;
  }
}

// ---- softmax(T) + floor + renorm mixing of 4 paths, RMS norm, f16 output
__global__ __launch_bounds__(256) void k_mix(const float* __restrict__ logits,
                                             const float* __restrict__ log_temp,
                                             const float* __restrict__ bb,
                                             const float* __restrict__ floor_raw,
                                             const float* __restrict__ p0,
                                             const float* __restrict__ p1,
                                             const float* __restrict__ p2,
                                             const float* __restrict__ p3,
                                             const float* __restrict__ onw,
                                             _Float16* __restrict__ mixed16) {
  int gw = blockIdx.x * 8 + (threadIdx.x >> 5);  // over M*H
  int lane = threadIdx.x & 31;
  int h = gw % H;
  int bl = gw / H;
  float temp = log1pf(expf(log_temp[h])) + 1e-4f;
  float lg[4], pr[4];
  float mx = -1e30f;
  #pragma unroll
  for (int i = 0; i < 4; ++i) {
    lg[i] = (logits[(size_t)bl * 16 + h * 4 + i] + bb[h * 4 + i]) / temp;
    mx = fmaxf(mx, lg[i]);
  }
  float s = 0.f;
  #pragma unroll
  for (int i = 0; i < 4; ++i) { pr[i] = expf(lg[i] - mx); s += pr[i]; }
  float s2 = 0.f;
  #pragma unroll
  for (int i = 0; i < 4; ++i) {
    pr[i] /= s;
    float fl = 0.05f / (1.f + expf(-floor_raw[h * 4 + i]));
    pr[i] = fmaxf(pr[i], fl);
    s2 += pr[i];
  }
  #pragma unroll
  for (int i = 0; i < 4; ++i) pr[i] /= s2;

  size_t baseo = (size_t)bl * D + h * DH;
  float acc[8];
  float ss = 0.f;
  #pragma unroll
  for (int j = 0; j < 8; ++j) {
    int d = j * 32 + lane;
    float o = pr[0] * p0[baseo + d] + pr[1] * p1[baseo + d] +
              pr[2] * p2[baseo + d] + pr[3] * p3[baseo + d];
    acc[j] = o; ss += o * o;
  }
  #pragma unroll
  for (int o = 16; o; o >>= 1) ss += __shfl_xor(ss, o);
  float scale = rsqrtf(ss / (float)DH + 1e-5f);
  #pragma unroll
  for (int j = 0; j < 8; ++j) {
    int d = j * 32 + lane;
    mixed16[baseo + d] = (_Float16)(acc[j] * scale * onw[d]);
  }
}

// =========================================================================
extern "C" void kernel_launch(void* const* d_in, const int* in_sizes, int n_in,
                              void* d_out, int out_size, void* d_ws, size_t ws_size,
                              hipStream_t stream) {
  const float* hs   = (const float*)d_in[0];
  const float* Wq   = (const float*)d_in[1];
  const float* Wk   = (const float*)d_in[2];
  const float* Wv   = (const float*)d_in[3];
  const float* Wb   = (const float*)d_in[4];
  const float* qcw  = (const float*)d_in[5];
  const float* kcw  = (const float*)d_in[6];
  const float* vcw  = (const float*)d_in[7];
  const float* fsw  = (const float*)d_in[8];
  const float* flw  = (const float*)d_in[9];
  const float* gw1  = (const float*)d_in[10];
  const float* gb1  = (const float*)d_in[11];
  const float* gw2  = (const float*)d_in[12];
  const float* ltmp = (const float*)d_in[13];
  const float* bb   = (const float*)d_in[14];
  const float* frw  = (const float*)d_in[15];
  const float* onw  = (const float*)d_in[16];
  const float* Wo   = (const float*)d_in[17];

  char* basep = (char*)d_ws;
  size_t off = 0;
  auto alloc = [&](size_t bytes) -> void* {
    off = (off + 255) & ~(size_t)255;
    void* p = basep + off;
    off += bytes;
    return p;
  };

  _Float16* h16    = (_Float16*)alloc((size_t)M * D * 2);
  _Float16* wq16   = (_Float16*)alloc((size_t)D * D * 2);
  _Float16* wk16   = (_Float16*)alloc((size_t)D * D * 2);
  _Float16* wv16   = (_Float16*)alloc((size_t)D * D * 2);
  _Float16* wo16   = (_Float16*)alloc((size_t)D * D * 2);
  _Float16* wg116  = (_Float16*)alloc((size_t)HID * GIN * 2);
  _Float16* wg216  = (_Float16*)alloc((size_t)16 * HID * 2);
  float*    qpre   = (float*)alloc((size_t)M * D * 4);
  float*    kpre   = (float*)alloc((size_t)M * D * 4);
  float*    vpre   = (float*)alloc((size_t)M * D * 4);
  float*    qc     = (float*)alloc((size_t)M * D * 4);
  float*    kc     = (float*)alloc((size_t)M * D * 4);
  float*    vc     = (float*)alloc((size_t)M * D * 4);
  float*    beta   = (float*)alloc((size_t)M * H * 4);
  _Float16* qn16   = (_Float16*)alloc((size_t)M * D * 2);
  _Float16* kn16   = (_Float16*)alloc((size_t)M * D * 2);
  _Float16* kb16   = (_Float16*)alloc((size_t)M * D * 2);
  _Float16* vb16   = (_Float16*)alloc((size_t)M * D * 2);
  float*    delta  = (float*)alloc((size_t)M * D * 4);
  float*    firs   = (float*)alloc((size_t)M * D * 4);
  float*    firl   = (float*)alloc((size_t)M * D * 4);
  float*    gin    = (float*)alloc((size_t)M * GIN * 4);
  _Float16* gin16  = (_Float16*)alloc((size_t)M * GIN * 2);
  float*    hmidr  = (float*)alloc((size_t)M * HID * 4);
  _Float16* hmid16 = (_Float16*)alloc((size_t)M * HID * 2);
  float*    logits = (float*)alloc((size_t)M * 16 * 4);
  _Float16* mix16  = (_Float16*)alloc((size_t)M * D * 2);

  auto gs = [](long long n) { return dim3((unsigned)((n + 255) / 256)); };

  // --- precision staging
  k_cvt_f16<<<gs((long long)M * D), 256, 0, stream>>>(hs, h16, (long long)M * D);
  k_cvt_f16<<<gs((long long)D * D), 256, 0, stream>>>(Wq, wq16, (long long)D * D);
  k_cvt_f16<<<gs((long long)D * D), 256, 0, stream>>>(Wk, wk16, (long long)D * D);
  k_cvt_f16<<<gs((long long)D * D), 256, 0, stream>>>(Wv, wv16, (long long)D * D);
  k_cvt_f16<<<gs((long long)D * D), 256, 0, stream>>>(Wo, wo16, (long long)D * D);
  k_cvt_f16<<<gs((long long)HID * GIN), 256, 0, stream>>>(gw1, wg116, (long long)HID * GIN);
  k_cvt_f16<<<gs((long long)16 * HID), 256, 0, stream>>>(gw2, wg216, (long long)16 * HID);

  // --- q/k/v projections (WMMA, 64x64 register tiles)
  dim3 gqkv(D / 64, M / 64);
  k_gemm64<<<gqkv, 32, 0, stream>>>(h16, wq16, qpre, M, D, D);
  k_gemm64<<<gqkv, 32, 0, stream>>>(h16, wk16, kpre, M, D, D);
  k_gemm64<<<gqkv, 32, 0, stream>>>(h16, wv16, vpre, M, D, D);

  // --- short conv + SiLU
  k_conv4_silu<<<gs((long long)M * D), 256, 0, stream>>>(qpre, qcw, qc);
  k_conv4_silu<<<gs((long long)M * D), 256, 0, stream>>>(kpre, kcw, kc);
  k_conv4_silu<<<gs((long long)M * D), 256, 0, stream>>>(vpre, vcw, vc);

  // --- beta, delta-rule preprocessing, delta scan
  k_beta<<<dim3(M * H / 8), 256, 0, stream>>>(hs, Wb, beta);
  k_prep<<<dim3(M * H / 8), 256, 0, stream>>>(qc, kc, vc, beta, qn16, kn16, kb16, vb16);
  k_delta<<<dim3(B * H, DH / 64), 128, 0, stream>>>(qn16, kn16, kb16, vb16, delta);

  // --- FIR paths
  k_fir<<<gs((long long)M * D), 256, 0, stream>>>(vc, fsw, firs, 3);
  k_fir<<<gs((long long)M * D), 256, 0, stream>>>(vc, flw, firl, 63);

  // --- gate input assembly
  k_copyhid<<<gs((long long)M * D), 256, 0, stream>>>(hs, gin, gin16);
  k_stats<<<dim3(M * 4 * H / 8), 256, 0, stream>>>(firs, firl, delta, vc, gin, gin16);

  // --- gate MLP (WMMA) + logits
  dim3 gmlp(HID / 64, M / 64);
  k_gemm64<<<gmlp, 32, 0, stream>>>(gin16, wg116, hmidr, M, HID, GIN);
  k_gelu<<<gs((long long)M * HID), 256, 0, stream>>>(hmidr, gb1, hmid16);
  dim3 glog(1, M / 16);
  k_gemm16<<<glog, 32, 0, stream>>>(hmid16, wg216, logits, M, 16, HID);

  // --- mix + RMS norm, then output projection (WMMA) straight into d_out
  k_mix<<<dim3(M * H / 8), 256, 0, stream>>>(logits, ltmp, bb, frw,
                                             firs, firl, delta, vc, onw, mix16);
  dim3 gout(D / 64, M / 64);
  k_gemm64<<<gout, 32, 0, stream>>>(mix16, wo16, (float*)d_out, M, D, D);
}
